// RelationalTemporalContrastiveLoss_33655363731900
// MI455X (gfx1250) — compile-verified, matrix operand order checked by reading, not run on hardware
//
#include <hip/hip_runtime.h>

#define N_PTS 8192
#define D_DIM 256
#define TEMP_INV (1.0f / 0.07f)
#define EPS_ALL 1e-8f
#define NCHUNK 8
#define TILES_PER_CHUNK (N_PTS / 16 / NCHUNK)   // 64 column tiles per chunk
// exp(sim) = exp2(dot * (1/T) * log2(e))
#define EXP2_SCALE (TEMP_INV * 1.44269504088896340736f)

typedef __attribute__((ext_vector_type(16))) __bf16 v16bf;
typedef __attribute__((ext_vector_type(8)))  float  v8f;
typedef unsigned short u16;
typedef unsigned int   u32;

// ---------------- helpers ----------------

__device__ __forceinline__ u16 f32_to_bf16_rne(float f) {
  u32 u = __float_as_uint(f);
  u32 r = u + 0x7FFFu + ((u >> 16) & 1u);
  return (u16)(r >> 16);
}

__device__ __forceinline__ float bf16_bits_to_f32(u16 b) {
  return __uint_as_float(((u32)b) << 16);
}

// ---------------- pass 1: normalize + hi/lo bf16 split ----------------
// zn = z / max(||z||, eps); zhi = bf16(zn); zlo = bf16(zn - f32(zhi))
__global__ __launch_bounds__(256) void prep_normalize_split(
    const float* __restrict__ z, u16* __restrict__ zhi, u16* __restrict__ zlo) {
  __shared__ float red[256];
  const int row = blockIdx.x;
  const int tid = threadIdx.x;
  float x = z[(size_t)row * D_DIM + tid];
  red[tid] = x * x;
  __syncthreads();
  for (int s = 128; s > 0; s >>= 1) {
    if (tid < s) red[tid] += red[tid + s];
    __syncthreads();
  }
  float norm   = sqrtf(red[0]);
  float scale  = 1.0f / fmaxf(norm, EPS_ALL);
  float xn     = x * scale;
  u16   hb     = f32_to_bf16_rne(xn);
  float hf     = bf16_bits_to_f32(hb);
  u16   lb     = f32_to_bf16_rne(xn - hf);
  zhi[(size_t)row * D_DIM + tid] = hb;
  zlo[(size_t)row * D_DIM + tid] = lb;
}

// ---------------- fragment loader ----------------
// 16-bit A-matrix 16x32 layout (ISA 7.12.2): lanes 0-15 hold K = kb+{0..7} in
// e0..e7 and K = kb+{16..23} in e8..e15; lanes 16-31 hold K = kb+{8..15} and
// kb+{24..31}.  B (32x16) of the symmetric GEMM uses the same per-lane gather
// with row = column index.  Two aligned 16-byte chunks per fragment.
union FragU { uint4 u[2]; v16bf v; };

__device__ __forceinline__ v16bf load_frag(const u16* __restrict__ base,
                                           int row, int kb, int hh) {
  const u16* p = base + (size_t)row * D_DIM + kb + hh * 8;
  FragU f;
  f.u[0] = *(const uint4*)(p);        // K = kb + hh*8 + 0..7
  f.u[1] = *(const uint4*)(p + 16);   // K = kb + 16 + hh*8 + 0..7
  return f.v;
}

__device__ __forceinline__ v8f wmma_bf16(v16bf a, v16bf b, v8f c) {
  return __builtin_amdgcn_wmma_f32_16x16x32_bf16(
      /*neg_a=*/false, a, /*neg_b=*/false, b,
      /*c_mod=*/(short)0, c, /*reuse_a=*/false, /*reuse_b=*/false);
}

// ---------------- pass 2: fused sim-GEMM + partial row sums ----------------
// grid = (128 row-blocks of 64, NCHUNK column chunks); block = 4 waves.
// Each wave owns a 16-row stripe and walks its chunk's 64 column tiles.
// sim*T^-1 recovered later; here we accumulate raw dots.
// sim_dot ~= Ahi*Bhi + Ahi*Blo + Alo*Bhi   (error-compensated bf16 split)
__global__ __launch_bounds__(128) void fused_partial(
    const u16*  __restrict__ zhi, const u16* __restrict__ zlo,
    const float* __restrict__ adj, const float* __restrict__ tsim,
    float* __restrict__ partials) {
  const int wave    = threadIdx.x >> 5;
  const int lane    = threadIdx.x & 31;
  const int hh      = lane >> 4;       // half-wave: 0 -> rows M+0..7, 1 -> M+8..15
  const int ll      = lane & 15;       // N index within tile (C layout)
  const int rowBase = blockIdx.x * 64 + wave * 16;
  const int chunk   = blockIdx.y;
  const int ct0     = chunk * TILES_PER_CHUNK;

  float accExp[8], accAS[8], accA[8], accTS[8], accT[8];
#pragma unroll
  for (int r = 0; r < 8; ++r) {
    accExp[r] = 0.f; accAS[r] = 0.f; accA[r] = 0.f; accTS[r] = 0.f; accT[r] = 0.f;
  }

  for (int t = 0; t < TILES_PER_CHUNK; ++t) {
    const int colBase = (ct0 + t) * 16;
    v8f c = {};
#pragma unroll
    for (int ks = 0; ks < 8; ++ks) {          // K = 256 in 8 steps of 32
      const int kb = ks * 32;
      v16bf ahi = load_frag(zhi, rowBase + ll, kb, hh);  // hoisted by compiler
      v16bf alo = load_frag(zlo, rowBase + ll, kb, hh);
      v16bf bhi = load_frag(zhi, colBase + ll, kb, hh);
      v16bf blo = load_frag(zlo, colBase + ll, kb, hh);
      c = wmma_bf16(ahi, bhi, c);
      c = wmma_bf16(ahi, blo, c);
      c = wmma_bf16(alo, bhi, c);
    }

    // C layout: VGPR r, this lane -> element (M = r + 8*hh, N = ll)
    const int j = colBase + ll;
    const float* adjp = adj  + (size_t)(rowBase + hh * 8) * N_PTS + j;
    const float* tsp  = tsim + (size_t)(rowBase + hh * 8) * N_PTS + j;

    if (colBase == rowBase) {                  // the one tile holding the diagonal
#pragma unroll
      for (int r = 0; r < 8; ++r) {
        const float dot = c[r];
        float e = exp2f(dot * EXP2_SCALE);
        if ((r + 8 * hh) == ll) e = 0.0f;      // mask diag in denominator only
        accExp[r] += e;
        const float a = __builtin_nontemporal_load(adjp + (size_t)r * N_PTS);
        const float w = __builtin_nontemporal_load(tsp  + (size_t)r * N_PTS);
        accA[r] += a;  accAS[r] += a * dot;
        accT[r] += w;  accTS[r] += w * dot;
      }
    } else {                                   // fast path: no per-element compare
#pragma unroll
      for (int r = 0; r < 8; ++r) {
        const float dot = c[r];
        accExp[r] += exp2f(dot * EXP2_SCALE);
        const float a = __builtin_nontemporal_load(adjp + (size_t)r * N_PTS);
        const float w = __builtin_nontemporal_load(tsp  + (size_t)r * N_PTS);
        accA[r] += a;  accAS[r] += a * dot;
        accT[r] += w;  accTS[r] += w * dot;
      }
    }
  }

  // Butterfly reduce across the 16 lanes of each half-wave (wave32).
#pragma unroll
  for (int r = 0; r < 8; ++r) {
    for (int m = 1; m < 16; m <<= 1) {
      accExp[r] += __shfl_xor(accExp[r], m, 32);
      accAS[r]  += __shfl_xor(accAS[r],  m, 32);
      accA[r]   += __shfl_xor(accA[r],   m, 32);
      accTS[r]  += __shfl_xor(accTS[r],  m, 32);
      accT[r]   += __shfl_xor(accT[r],   m, 32);
    }
  }

  if (ll == 0) {
    // partials layout: [chunk][stat 0..4][row]  -> coalesced in finalize
#pragma unroll
    for (int r = 0; r < 8; ++r) {
      const int i = rowBase + hh * 8 + r;
      partials[((size_t)chunk * 5 + 0) * N_PTS + i] = accExp[r];
      partials[((size_t)chunk * 5 + 1) * N_PTS + i] = accAS[r];
      partials[((size_t)chunk * 5 + 2) * N_PTS + i] = accA[r];
      partials[((size_t)chunk * 5 + 3) * N_PTS + i] = accTS[r];
      partials[((size_t)chunk * 5 + 4) * N_PTS + i] = accT[r];
    }
  }
}

// ---------------- pass 3: combine chunks, per-row loss ----------------
__global__ __launch_bounds__(256) void finalize_rows(
    const float* __restrict__ partials, float* __restrict__ row_loss) {
  const int i = blockIdx.x * 256 + threadIdx.x;
  float E = 0.f, AS = 0.f, A = 0.f, TS = 0.f, T = 0.f;
#pragma unroll
  for (int c = 0; c < NCHUNK; ++c) {
    E  += partials[((size_t)c * 5 + 0) * N_PTS + i];
    AS += partials[((size_t)c * 5 + 1) * N_PTS + i];
    A  += partials[((size_t)c * 5 + 2) * N_PTS + i];
    TS += partials[((size_t)c * 5 + 3) * N_PTS + i];
    T  += partials[((size_t)c * 5 + 4) * N_PTS + i];
  }
  AS *= TEMP_INV;  // raw dots -> sim
  TS *= TEMP_INV;
  const float logden = __logf(E + EPS_ALL);
  // -(w*log_prob).sum / (w.sum+eps) = (logden*Σw - Σ(w*sim)) / (Σw+eps)
  const float lrel = (logden * A - AS) / (A + EPS_ALL);
  const float ltmp = (logden * T - TS) / (T + EPS_ALL);
  row_loss[i] = (lrel + ltmp) * (1.0f / (float)N_PTS);
}

// ---------------- pass 4: deterministic scalar reduction ----------------
__global__ __launch_bounds__(256) void reduce_rows(
    const float* __restrict__ row_loss, float* __restrict__ out) {
  __shared__ float red[256];
  float s = 0.f;
  for (int i = threadIdx.x; i < N_PTS; i += 256) s += row_loss[i];
  red[threadIdx.x] = s;
  __syncthreads();
  for (int k = 128; k > 0; k >>= 1) {
    if (threadIdx.x < k) red[threadIdx.x] += red[threadIdx.x + k];
    __syncthreads();
  }
  if (threadIdx.x == 0) out[0] = red[0];
}

// ---------------- launch ----------------
extern "C" void kernel_launch(void* const* d_in, const int* in_sizes, int n_in,
                              void* d_out, int out_size, void* d_ws, size_t ws_size,
                              hipStream_t stream) {
  const float* z    = (const float*)d_in[0];   // [8192, 256]
  const float* adj  = (const float*)d_in[1];   // [8192, 8192]
  const float* tsim = (const float*)d_in[2];   // [8192, 8192]
  float* out = (float*)d_out;

  // workspace: zn_hi (4MB) | zn_lo (4MB) | partials (1.25MB) | row_loss (32KB)
  u16*   zhi      = (u16*)d_ws;
  u16*   zlo      = zhi + (size_t)N_PTS * D_DIM;
  float* partials = (float*)(zlo + (size_t)N_PTS * D_DIM);
  float* row_loss = partials + (size_t)NCHUNK * 5 * N_PTS;

  prep_normalize_split<<<N_PTS, 256, 0, stream>>>(z, zhi, zlo);
  dim3 grid(N_PTS / 64, NCHUNK);
  fused_partial<<<grid, 128, 0, stream>>>(zhi, zlo, adj, tsim, partials);
  finalize_rows<<<N_PTS / 256, 256, 0, stream>>>(partials, row_loss);
  reduce_rows<<<1, 256, 0, stream>>>(row_loss, out);
}
